// AttentionTypeEnsembleSheafLearner_31842887533262
// MI455X (gfx1250) — compile-verified, matrix operand order checked by reading, not run on hardware
//
#include <hip/hip_runtime.h>
#include <hip/hip_bf16.h>

typedef __attribute__((ext_vector_type(16))) _Float16 v16h;
typedef __attribute__((ext_vector_type(8)))  float    v8f;

#define T_TYPES 8
#define WPB 4              // waves per block (wave32)
#define TPB (WPB * 32)

union AF { unsigned int u[8]; v16h v; };

__device__ __forceinline__ v8f zero8() {
    v8f z;
#pragma unroll
    for (int i = 0; i < 8; ++i) z[i] = 0.0f;
    return z;
}

__device__ __forceinline__ unsigned int pack2(float a, float b) {
    union { _Float16 h[2]; unsigned int u; } cv;
    cv.h[0] = (_Float16)a; cv.h[1] = (_Float16)b;
    return cv.u;
}

// K index inside a 16-bit WMMA fragment for (vgpr v, lane-half hlf)
__device__ __forceinline__ int fragK(int v, int hlf) {
    return 2 * (v & 3) + 16 * (v >> 2) + 8 * hlf;
}

// ---------- binning kernels ----------
__global__ void k_init(int* ws) {
    if (threadIdx.x < 24) ws[threadIdx.x] = 0;   // cnt[8], offs[8], cursor[8]
}

__global__ void k_count(const long long* __restrict__ et, int* __restrict__ cnt, int E) {
    int e = blockIdx.x * blockDim.x + threadIdx.x;
    if (e < E) atomicAdd(&cnt[(int)et[e]], 1);
}

__global__ void k_scan(const int* __restrict__ cnt, int* __restrict__ offs) {
    int acc = 0;
    for (int t = 0; t < T_TYPES; ++t) { offs[t] = acc; acc += cnt[t]; }
}

__global__ void k_scatter(const long long* __restrict__ et, int* __restrict__ cur,
                          const int* __restrict__ offs, int* __restrict__ perm, int E) {
    int e = blockIdx.x * blockDim.x + threadIdx.x;
    if (e < E) {
        int t = (int)et[e];
        int slot = atomicAdd(&cur[t], 1);
        perm[offs[t] + slot] = e;
    }
}

// ---------- main fused kernel ----------
// One 16-edge, type-uniform tile per wave per iteration.
// Layer1: [16x256]x[256x64] = 32x wmma_f32_16x16x32_f16
// Layer2: [16x64]x[64x16]   =  2x wmma_f32_16x16x32_f16
// sW1/sW2 stored in exact fragment order -> every B frag is 2x ds_load_b128.
// Next tile's node rows are prefetched (global_prefetch_b8) at the end of the
// gather phase so the prefetch address regs die before the WMMA region.
__global__ __launch_bounds__(TPB)
void sheaf_main(const float* __restrict__ x, const long long* __restrict__ ei,
                const float* __restrict__ gamma, const float* __restrict__ beta,
                const float* __restrict__ W1, const float* __restrict__ b1,
                const float* __restrict__ W2, const float* __restrict__ b2,
                const int* __restrict__ perm, const int* __restrict__ cnt,
                const int* __restrict__ offs, float* __restrict__ out, int Etot)
{
    // sW1: frag-ordered [(kk*4+nt)*32 + lane]*8 + v   (8 kk x 4 nt x 32 lanes x 8) = 32KB
    __shared__ unsigned int sW1[8 * 4 * 32 * 8];
    // sW2: frag-ordered [kk*32 + lane]*8 + v          (2 kk x 32 lanes x 8) = 2KB
    __shared__ unsigned int sW2[2 * 32 * 8];
    __shared__ float sGamma[256], sBeta[256], sB1[64], sB2[16];
    __shared__ unsigned int sA[WPB][16 * 128]; // per-wave A tile (reused as H)

    const int t    = blockIdx.y;
    const int tid  = threadIdx.x;
    const int lane = tid & 31;
    const int wave = tid >> 5;

    // ---- stage per-type params (fragment-order swizzle for W1/W2) ----
    const float* W1g = W1 + (size_t)t * 256 * 64;
    for (int i = tid; i < 8 * 4 * 32 * 8; i += TPB) {
        int v  = i & 7;
        int ln = (i >> 3) & 31;
        int nt = (i >> 8) & 3;
        int kk = (i >> 10) & 7;
        int K  = kk * 32 + fragK(v, ln >> 4);
        int n  = nt * 16 + (ln & 15);
        sW1[i] = pack2(W1g[K * 64 + n], W1g[(K + 1) * 64 + n]);
    }
    const float* W2g = W2 + (size_t)t * 64 * 16;
    for (int i = tid; i < 2 * 32 * 8; i += TPB) {
        int v  = i & 7;
        int ln = (i >> 3) & 31;
        int kk = (i >> 8) & 1;
        int K  = kk * 32 + fragK(v, ln >> 4);
        int n  = ln & 15;
        sW2[i] = pack2(W2g[K * 16 + n], W2g[(K + 1) * 16 + n]);
    }
    for (int i = tid; i < 256; i += TPB) { sGamma[i] = gamma[t * 256 + i]; sBeta[i] = beta[t * 256 + i]; }
    for (int i = tid; i < 64;  i += TPB) sB1[i] = b1[t * 64 + i];
    for (int i = tid; i < 16;  i += TPB) sB2[i] = b2[t * 16 + i];
    __syncthreads();

    const int count      = cnt[t];
    const int start      = offs[t];
    const int numTiles   = (count + 15) >> 4;
    const int waveStride = gridDim.x * WPB;
    const int iters      = (numTiles + waveStride - 1) / waveStride;

    unsigned int* myA = sA[wave];
    _Float16*     myH = (_Float16*)myA;

    const int m16 = lane & 15;     // N (or M) index within 16
    const int hlf = lane >> 4;     // lane-half selects K sub-range / M+8

    for (int it = 0; it < iters; ++it) {
        const int tile = it * waveStride + blockIdx.x * WPB + wave;
        const bool active = tile < numTiles;

        if (active) {
            // ---- gather (kept in registers) + shared layernorm + affine -> f16 A ----
            const int m    = lane >> 1;        // edge within tile
            const int half = lane & 1;         // 0: row endpoint, 1: col endpoint
            const int eSlot = tile * 16 + m;
            const int eIdx  = (eSlot < count) ? perm[start + eSlot] : perm[start];
            const long long node = (half == 0) ? ei[eIdx] : ei[Etot + eIdx];
            const float4* xr = (const float4*)(x + (size_t)node * 128);

            float4 buf[32];
            float s = 0.f, ss = 0.f;
#pragma unroll
            for (int j = 0; j < 32; ++j) {
                float4 v = xr[j];
                buf[j] = v;
                s  += v.x + v.y + v.z + v.w;
                ss += v.x * v.x + v.y * v.y + v.z * v.z + v.w * v.w;
            }
            s  += __shfl_xor(s, 1, 32);
            ss += __shfl_xor(ss, 1, 32);
            const float mean = s * (1.f / 256.f);
            const float var  = ss * (1.f / 256.f) - mean * mean;
            const float rstd = rsqrtf(var + 1e-5f);

            const int cbase = half * 128;
#pragma unroll
            for (int j = 0; j < 32; ++j) {
                float4 v = buf[j];
                int c = cbase + 4 * j;
                float f0 = (v.x - mean) * rstd * sGamma[c + 0] + sBeta[c + 0];
                float f1 = (v.y - mean) * rstd * sGamma[c + 1] + sBeta[c + 1];
                float f2 = (v.z - mean) * rstd * sGamma[c + 2] + sBeta[c + 2];
                float f3 = (v.w - mean) * rstd * sGamma[c + 3] + sBeta[c + 3];
                myA[m * 128 + (c >> 1)]     = pack2(f0, f1);
                myA[m * 128 + (c >> 1) + 1] = pack2(f2, f3);
            }

            // ---- prefetch next tile's node rows; fire-and-forget, regs die here ----
            const int ntile = tile + waveStride;
            if (ntile < numTiles) {
                int peSlot = ntile * 16 + m;
                int peIdx  = (peSlot < count) ? perm[start + peSlot] : perm[start];
                long long pnode = (half == 0) ? ei[peIdx] : ei[Etot + peIdx];
                const char* p = (const char*)(x + (size_t)pnode * 128);
                __builtin_prefetch(p, 0, 3);          // global_prefetch_b8
                __builtin_prefetch(p + 128, 0, 3);
                __builtin_prefetch(p + 256, 0, 3);
                __builtin_prefetch(p + 384, 0, 3);
            }
        }
        __syncthreads();

        if (active) {
            // ---- layer 1: kk-outer / nt-inner keeps live regs ~50 ----
            v8f acc[4];
#pragma unroll
            for (int nt = 0; nt < 4; ++nt) acc[nt] = zero8();
#pragma unroll
            for (int kk = 0; kk < 8; ++kk) {
                AF a;
                const int abase = m16 * 128 + kk * 16 + 4 * hlf;
#pragma unroll
                for (int v = 0; v < 4; ++v) a.u[v] = myA[abase + v];          // ds_load_b128
#pragma unroll
                for (int v = 0; v < 4; ++v) a.u[4 + v] = myA[abase + 8 + v];  // ds_load_b128
#pragma unroll
                for (int nt = 0; nt < 4; ++nt) {
                    AF bf;
                    const int bbase = ((kk * 4 + nt) * 32 + lane) * 8;
#pragma unroll
                    for (int v = 0; v < 8; ++v) bf.u[v] = sW1[bbase + v];     // 2x ds_load_b128
                    acc[nt] = __builtin_amdgcn_wmma_f32_16x16x32_f16(
                        false, a.v, false, bf.v, (short)0, acc[nt], false, false);
                }
            }
            // ReLU + b1, re-layout C->A via LDS (A buffer fully consumed above)
#pragma unroll
            for (int nt = 0; nt < 4; ++nt)
#pragma unroll
                for (int i = 0; i < 8; ++i) {
                    int mm = i + 8 * hlf;
                    int nn = nt * 16 + m16;
                    float vl = acc[nt][i] + sB1[nn];
                    myH[mm * 64 + nn] = (_Float16)(vl > 0.f ? vl : 0.f);
                }
        }
        __syncthreads();

        if (active) {
            // ---- layer 2: H[16x64] x W2[64x16] ----
            v8f acc2 = zero8();
#pragma unroll
            for (int kk = 0; kk < 2; ++kk) {
                AF a2, b2f;
                const int abase = m16 * 32 + kk * 16 + 4 * hlf;
#pragma unroll
                for (int v = 0; v < 4; ++v) a2.u[v] = myA[abase + v];
#pragma unroll
                for (int v = 0; v < 4; ++v) a2.u[4 + v] = myA[abase + 8 + v];
                const int bbase = (kk * 32 + lane) * 8;
#pragma unroll
                for (int v = 0; v < 8; ++v) b2f.u[v] = sW2[bbase + v];
                acc2 = __builtin_amdgcn_wmma_f32_16x16x32_f16(
                    false, a2.v, false, b2f.v, (short)0, acc2, false, false);
            }
            // ---- bias + row softmax over groups of 4 + (I - attn) + store ----
            const int n    = m16;
            const int jcol = n & 3;
            const int irow = n >> 2;
            const float id = (irow == jcol) ? 1.f : 0.f;
#pragma unroll
            for (int i = 0; i < 8; ++i) {
                int   mm = i + 8 * hlf;
                float vv = acc2[i] + sB2[n];
                float mx = fmaxf(vv, __shfl_xor(vv, 1, 32));
                mx       = fmaxf(mx, __shfl_xor(mx, 2, 32));
                float ex = __expf(vv - mx);
                float sm = ex + __shfl_xor(ex, 1, 32);
                sm      += __shfl_xor(sm, 2, 32);
                float attn = ex / sm;
                int eSlot = tile * 16 + mm;
                if (eSlot < count) {
                    int e = perm[start + eSlot];
                    out[(size_t)e * 16 + n] = id - attn;
                }
            }
        }
        __syncthreads();
    }
}

extern "C" void kernel_launch(void* const* d_in, const int* in_sizes, int n_in,
                              void* d_out, int out_size, void* d_ws, size_t ws_size,
                              hipStream_t stream) {
    const float*     x     = (const float*)d_in[0];
    const long long* ei    = (const long long*)d_in[1];
    const long long* et    = (const long long*)d_in[2];
    const float*     gamma = (const float*)d_in[3];
    const float*     beta  = (const float*)d_in[4];
    const float*     W1    = (const float*)d_in[5];
    const float*     b1    = (const float*)d_in[6];
    const float*     W2    = (const float*)d_in[7];
    const float*     b2    = (const float*)d_in[8];
    float*           out   = (float*)d_out;
    const int E = in_sizes[2];

    int* ws   = (int*)d_ws;
    int* cnt  = ws;        // [8]
    int* offs = ws + 8;    // [8]
    int* cur  = ws + 16;   // [8]
    int* perm = ws + 32;   // [E]

    k_init<<<1, 32, 0, stream>>>(ws);
    k_count<<<(E + 255) / 256, 256, 0, stream>>>(et, cnt, E);
    k_scan<<<1, 1, 0, stream>>>(cnt, offs);
    k_scatter<<<(E + 255) / 256, 256, 0, stream>>>(et, cur, offs, perm, E);

    dim3 grid(64, T_TYPES);
    sheaf_main<<<grid, TPB, 0, stream>>>(x, ei, gamma, beta, W1, b1, W2, b2,
                                         perm, cnt, offs, out, E);
}
